// TransformerEncoderLayerRPR_5909875000076
// MI455X (gfx1250) — matrix operand order repaired and checked by reading, not compiled, not run on hardware
//
#include <hip/hip_runtime.h>

typedef __attribute__((ext_vector_type(16))) __bf16 v16bf;
typedef __attribute__((ext_vector_type(8)))  float  v8f;

union FragB16 { v16bf v; uint4 q[2]; unsigned short s[16]; };
union Acc8    { v8f v; float f[8]; };

__device__ __forceinline__ unsigned short f2bf(float x) {
  unsigned u = __float_as_uint(x);
  u += 0x7fffu + ((u >> 16) & 1u);
  return (unsigned short)(u >> 16);
}

// ---------------------------------------------------------------- converts
__global__ void cvt_f32_to_bf16(const float* __restrict__ in,
                                unsigned short* __restrict__ out, long long n) {
  long long i = (long long)blockIdx.x * blockDim.x + threadIdx.x;
  long long stride = (long long)gridDim.x * blockDim.x;
  for (; i < n; i += stride) out[i] = f2bf(in[i]);
}

// QKV bf16 (S*2, 3072) -> Vt bf16 (32 heads, 64, 2048)  (K-major V for WMMA B-frags)
__global__ void v_transpose(const unsigned short* __restrict__ QKV,
                            unsigned short* __restrict__ Vt) {
  int tid = blockIdx.x * blockDim.x + threadIdx.x;  // 32*64*2048 total
  int s = tid & 2047;
  int d = (tid >> 11) & 63;
  int g = tid >> 17;
  int b = g >> 4, hh = g & 15;
  Vt[tid] = QKV[((size_t)s * 2 + b) * 3072 + 2048 + (size_t)hh * 64 + d];
}

// ---------------------------------------------------------------- GEMM
// C[row,col] = act( (sum_k A[row,k]*W[col,k] + bias[col]) * scale_if(col<limit) )
// A:(rows,K) bf16 row-major, W:(N,K) bf16 row-major.
// Wave computes 32x64; block (8 waves) = 128x128. Register double-buffered k-loop.
__global__ __launch_bounds__(256)
void gemm_bf16_wmma(const unsigned short* __restrict__ A,
                    const unsigned short* __restrict__ W,
                    const float* __restrict__ bias,
                    float* __restrict__ outF,
                    unsigned short* __restrict__ outB,
                    int K, int N, float scaleVal, int scaleLimit, int doRelu) {
  const int lane = threadIdx.x & 31;
  const int wave = threadIdx.x >> 5;
  const int rowBase = blockIdx.y * 128 + (wave >> 1) * 32;
  const int colBase = blockIdx.x * 128 + (wave & 1) * 64;
  const int l15 = lane & 15;
  const int hA = (lane & 16) ? 8 : 0;    // A-frag K base
  const int hB = (lane & 16) ? 16 : 0;   // unused here (B-frags use hA)
  const int mrow = (lane & 16) ? 8 : 0;  // C row base

  const unsigned short* Ar0 = A + (size_t)(rowBase + l15) * K + hA;
  const unsigned short* Ar1 = A + (size_t)(rowBase + 16 + l15) * K + hA;
  const unsigned short* Wp0 = W + (size_t)(colBase + l15) * K + hA;

  Acc8 acc[2][4];
#pragma unroll
  for (int rt = 0; rt < 2; ++rt)
#pragma unroll
    for (int t = 0; t < 4; ++t)
#pragma unroll
      for (int r = 0; r < 8; ++r) acc[rt][t].f[r] = 0.f;

  auto loadA = [&](FragB16* dst, int k0) {
    dst[0].q[0] = *(const uint4*)(Ar0 + k0);
    dst[0].q[1] = *(const uint4*)(Ar0 + k0 + 16);
    dst[1].q[0] = *(const uint4*)(Ar1 + k0);
    dst[1].q[1] = *(const uint4*)(Ar1 + k0 + 16);
  };
  auto loadB = [&](FragB16* dst, int k0) {
#pragma unroll
    for (int t = 0; t < 4; ++t) {
      const unsigned short* wp = Wp0 + (size_t)t * 16 * K + k0;
      dst[t].q[0] = *(const uint4*)(wp);
      dst[t].q[1] = *(const uint4*)(wp + 16);
    }
  };
  auto mmaStep = [&](FragB16* a, FragB16* b) {
#pragma unroll
    for (int t = 0; t < 4; ++t) {
      acc[0][t].v = __builtin_amdgcn_wmma_f32_16x16x32_bf16(
          false, a[0].v, false, b[t].v, (short)0, acc[0][t].v, false, false);
      acc[1][t].v = __builtin_amdgcn_wmma_f32_16x16x32_bf16(
          false, a[1].v, false, b[t].v, (short)0, acc[1][t].v, false, false);
    }
  };

  FragB16 a0[2], a1[2], b0[4], b1[4];
  loadA(a0, 0);
  loadB(b0, 0);
  for (int k0 = 0; k0 < K; k0 += 64) {  // K % 64 == 0 for all our GEMMs
    // issue next-chunk loads before consuming current chunk
    loadA(a1, k0 + 32);
    loadB(b1, k0 + 32);
    __builtin_prefetch(Ar0 + k0 + 64, 0, 1);  // global_prefetch_b8
    mmaStep(a0, b0);
    if (k0 + 64 < K) {
      loadA(a0, k0 + 64);
      loadB(b0, k0 + 64);
    }
    mmaStep(a1, b1);
  }

#pragma unroll
  for (int rt = 0; rt < 2; ++rt) {
#pragma unroll
    for (int t = 0; t < 4; ++t) {
      const int col = colBase + t * 16 + l15;
      const float bv = bias ? bias[col] : 0.f;
      const float sc = (col < scaleLimit) ? scaleVal : 1.f;
#pragma unroll
      for (int r = 0; r < 8; ++r) {
        const int row = rowBase + rt * 16 + mrow + r;
        float v = (acc[rt][t].f[r] + bv) * sc;
        if (doRelu) v = fmaxf(v, 0.f);
        const size_t o = (size_t)row * N + col;
        if (outF) outF[o] = v;
        if (outB) outB[o] = f2bf(v);
      }
    }
  }
}

// ---------------------------------------------------------------- attention
#define ATT_S 2048
// scores[i,j] = q_i.k_j + (j<=i ? q_i.er[S-1+j-i] : 0); softmax; @V. q pre-scaled.
__global__ __launch_bounds__(32)
void attn_rpr_flash(const unsigned short* __restrict__ QKV,  // (S*2, 3072) bf16
                    const unsigned short* __restrict__ Er,   // (S, 64)     bf16
                    const unsigned short* __restrict__ Vt,   // (32, 64, S) bf16
                    unsigned short* __restrict__ AO) {       // (S*2, 1024) bf16
  __shared__ float lds[16 * 32];
  const int i0 = blockIdx.x * 16;
  const int gh = blockIdx.y;  // head 0..31 = b*16+hh
  const int b = gh >> 4, hh = gh & 15;
  const int lane = threadIdx.x;
  const int l15 = lane & 15;
  const int hA = (lane & 16) ? 8 : 0;
  const int hB = (lane & 16) ? 16 : 0;
  const int mrow = (lane & 16) ? 8 : 0;

  FragB16 qf[2];
  {
    const size_t qb = ((size_t)(i0 + l15) * 2 + b) * 3072 + (size_t)hh * 64;
#pragma unroll
    for (int kc = 0; kc < 2; ++kc) {
      qf[kc].q[0] = *(const uint4*)(QKV + qb + kc * 32 + hB);
      qf[kc].q[1] = *(const uint4*)(QKV + qb + kc * 32 + hB + 8);
    }
  }

  float runM[8], runL[8];
  Acc8 Ov[4];
#pragma unroll
  for (int r = 0; r < 8; ++r) { runM[r] = -3.0e38f; runL[r] = 0.f; }
#pragma unroll
  for (int t = 0; t < 4; ++t)
#pragma unroll
    for (int r = 0; r < 8; ++r) Ov[t].f[r] = 0.f;

  const unsigned short* Vg = Vt + (size_t)gh * 64 * ATT_S;

  for (int j0 = 0; j0 < ATT_S; j0 += 32) {
    // ---- issue all K/V fragment loads for this chunk up front ----
    FragB16 kf[2][2], vf[4];
#pragma unroll
    for (int sub = 0; sub < 2; ++sub)
#pragma unroll
      for (int kc = 0; kc < 2; ++kc) {
        const size_t kb = ((size_t)(j0 + sub * 16 + l15) * 2 + b) * 3072 + 1024 +
                          (size_t)hh * 64 + kc * 32 + hB;
        kf[sub][kc].q[0] = *(const uint4*)(QKV + kb);
        kf[sub][kc].q[1] = *(const uint4*)(QKV + kb + 8);
      }
#pragma unroll
    for (int t = 0; t < 4; ++t) {
      const unsigned short* vp = Vg + (size_t)(t * 16 + l15) * ATT_S + j0 + hB;
      vf[t].q[0] = *(const uint4*)(vp);
      vf[t].q[1] = *(const uint4*)(vp + 8);
    }

    float srow[2][8], p0s[8], p1s[8];
#pragma unroll
    for (int sub = 0; sub < 2; ++sub) {
      const int j0s = j0 + sub * 16;
      Acc8 sc;
#pragma unroll
      for (int r = 0; r < 8; ++r) sc.f[r] = 0.f;
#pragma unroll
      for (int kc = 0; kc < 2; ++kc)
        sc.v = __builtin_amdgcn_wmma_f32_16x16x32_bf16(
            false, qf[kc].v, false, kf[sub][kc].v, (short)0, sc.v, false, false);

      const bool needRel = (j0s <= i0 + 15);  // wave-uniform
      if (needRel) {
        const int c0 = ATT_S - 1 - i0 + j0s;  // m = c0 + j' - i'
        Acc8 qe[2];
#pragma unroll
        for (int cb = 0; cb < 2; ++cb) {
#pragma unroll
          for (int r = 0; r < 8; ++r) qe[cb].f[r] = 0.f;
#pragma unroll
          for (int kc = 0; kc < 2; ++kc) {
            const int m = c0 - 15 + cb * 16 + l15;
            FragB16 ef;
            if (m >= 0 && m < ATT_S) {
              const unsigned short* ep = Er + (size_t)m * 64 + kc * 32 + hB;
              ef.q[0] = *(const uint4*)(ep);
              ef.q[1] = *(const uint4*)(ep + 8);
            } else {
              ef.q[0] = make_uint4(0, 0, 0, 0);
              ef.q[1] = make_uint4(0, 0, 0, 0);
            }
            qe[cb].v = __builtin_amdgcn_wmma_f32_16x16x32_bf16(
                false, qf[kc].v, false, ef.v, (short)0, qe[cb].v, false, false);
          }
        }
        __syncthreads();
#pragma unroll
        for (int cb = 0; cb < 2; ++cb)
#pragma unroll
          for (int r = 0; r < 8; ++r)
            lds[(mrow + r) * 32 + cb * 16 + l15] = qe[cb].f[r];
        __syncthreads();
      }
#pragma unroll
      for (int r = 0; r < 8; ++r) {
        const int M = mrow + r;
        float rel = 0.f;
        if (needRel && (j0s + l15) <= (i0 + M)) rel = lds[M * 32 + 15 + l15 - M];
        srow[sub][r] = sc.f[r] + rel;
      }
    }
    // online softmax over these 32 keys (row = 16 lanes of one half)
#pragma unroll
    for (int r = 0; r < 8; ++r) {
      float mx = fmaxf(srow[0][r], srow[1][r]);
#pragma unroll
      for (int off = 8; off > 0; off >>= 1) mx = fmaxf(mx, __shfl_xor(mx, off));
      const float newM = fmaxf(runM[r], mx);
      const float alpha = __expf(runM[r] - newM);
      const float p0 = __expf(srow[0][r] - newM);
      const float p1 = __expf(srow[1][r] - newM);
      float rs = p0 + p1;
#pragma unroll
      for (int off = 8; off > 0; off >>= 1) rs += __shfl_xor(rs, off);
      runL[r] = runL[r] * alpha + rs;
      runM[r] = newM;
      p0s[r] = p0; p1s[r] = p1;
#pragma unroll
      for (int t = 0; t < 4; ++t) Ov[t].f[r] *= alpha;
    }
    // transpose P (16x32) through LDS into A-fragment order
    __syncthreads();
#pragma unroll
    for (int r = 0; r < 8; ++r) {
      lds[(mrow + r) * 32 + l15]      = p0s[r];
      lds[(mrow + r) * 32 + 16 + l15] = p1s[r];
    }
    __syncthreads();
    FragB16 pa;
#pragma unroll
    for (int e = 0; e < 16; ++e) {
      const int kk = hB + e;
      pa.s[e] = f2bf(lds[l15 * 32 + kk]);
    }
#pragma unroll
    for (int t = 0; t < 4; ++t)
      Ov[t].v = __builtin_amdgcn_wmma_f32_16x16x32_bf16(
          false, pa.v, false, vf[t].v, (short)0, Ov[t].v, false, false);
  }

#pragma unroll
  for (int r = 0; r < 8; ++r) {
    const float inv = 1.f / runL[r];
    const size_t ob = ((size_t)(i0 + mrow + r) * 2 + b) * 1024 + (size_t)hh * 64;
#pragma unroll
    for (int t = 0; t < 4; ++t) AO[ob + t * 16 + l15] = f2bf(Ov[t].f[r] * inv);
  }
}

// ---------------------------------------------------------------- add + LN
__global__ __launch_bounds__(256)
void add_layernorm(const float* __restrict__ a, const float* __restrict__ b,
                   const float* __restrict__ g, const float* __restrict__ beta,
                   float* __restrict__ outF, unsigned short* __restrict__ outB,
                   int E) {
  const int row = blockIdx.x;
  const float* ar = a + (size_t)row * E;
  const float* br = b + (size_t)row * E;
  float s = 0.f, s2 = 0.f;
  for (int i = threadIdx.x; i < E; i += blockDim.x) {
    float v = ar[i] + br[i];
    s += v; s2 += v * v;
  }
#pragma unroll
  for (int off = 16; off > 0; off >>= 1) {
    s  += __shfl_xor(s, off);
    s2 += __shfl_xor(s2, off);
  }
  __shared__ float red[2][8];
  const int wid = threadIdx.x >> 5;
  if ((threadIdx.x & 31) == 0) { red[0][wid] = s; red[1][wid] = s2; }
  __syncthreads();
  float ts = 0.f, ts2 = 0.f;
#pragma unroll
  for (int w = 0; w < 8; ++w) { ts += red[0][w]; ts2 += red[1][w]; }
  const float mean = ts / E;
  const float var  = ts2 / E - mean * mean;
  const float inv  = rsqrtf(var + 1e-5f);
  for (int i = threadIdx.x; i < E; i += blockDim.x) {
    float v = (ar[i] + br[i] - mean) * inv * g[i] + beta[i];
    const size_t o = (size_t)row * E + i;
    if (outF) outF[o] = v;
    if (outB) outB[o] = f2bf(v);
  }
}

// ---------------------------------------------------------------- driver
extern "C" void kernel_launch(void* const* d_in, const int* in_sizes, int n_in,
                              void* d_out, int out_size, void* d_ws, size_t ws_size,
                              hipStream_t stream) {
  (void)in_sizes; (void)n_in; (void)out_size; (void)ws_size;
  const float* src = (const float*)d_in[0];
  const float* ipw = (const float*)d_in[1];
  const float* ipb = (const float*)d_in[2];
  const float* opw = (const float*)d_in[3];
  const float* opb = (const float*)d_in[4];
  const float* er  = (const float*)d_in[5];
  const float* w1  = (const float*)d_in[6];
  const float* b1  = (const float*)d_in[7];
  const float* w2  = (const float*)d_in[8];
  const float* b2  = (const float*)d_in[9];
  const float* g1  = (const float*)d_in[10];
  const float* be1 = (const float*)d_in[11];
  const float* g2  = (const float*)d_in[12];
  const float* be2 = (const float*)d_in[13];
  float* out = (float*)d_out;

  const int S = 2048, Bk = 2, E = 1024, F = 4096;
  const int NR = S * Bk;  // 4096 GEMM rows

  char* p = (char*)d_ws;
  auto take = [&](size_t bytes) {
    char* r = p;
    p += (bytes + 255) & ~(size_t)255;
    return r;
  };
  unsigned short* Wqkv = (unsigned short*)take((size_t)3 * E * E * 2);
  unsigned short* Wout = (unsigned short*)take((size_t)E * E * 2);
  unsigned short* W1b  = (unsigned short*)take((size_t)F * E * 2);
  unsigned short* W2b  = (unsigned short*)take((size_t)E * F * 2);
  unsigned short* Erb  = (unsigned short*)take((size_t)S * 64 * 2);
  unsigned short* X0b  = (unsigned short*)take((size_t)NR * E * 2);
  unsigned short* QKVb = (unsigned short*)take((size_t)NR * 3 * E * 2);
  unsigned short* Vtb  = (unsigned short*)take((size_t)32 * 64 * S * 2);
  unsigned short* AOb  = (unsigned short*)take((size_t)NR * E * 2);
  float*          PROJ = (float*)take((size_t)NR * E * 4);
  float*          X1f  = (float*)take((size_t)NR * E * 4);
  unsigned short* X1b  = (unsigned short*)take((size_t)NR * E * 2);
  unsigned short* Hb   = (unsigned short*)take((size_t)NR * F * 2);
  float*          FFf  = (float*)take((size_t)NR * E * 4);

  auto cvt = [&](const float* in, unsigned short* o, long long n) {
    int blocks = (int)((n + 255) / 256);
    if (blocks > 4096) blocks = 4096;
    cvt_f32_to_bf16<<<dim3(blocks), dim3(256), 0, stream>>>(in, o, n);
  };
  cvt(src, X0b, (long long)NR * E);
  cvt(ipw, Wqkv, (long long)3 * E * E);
  cvt(opw, Wout, (long long)E * E);
  cvt(w1, W1b, (long long)F * E);
  cvt(w2, W2b, (long long)E * F);
  cvt(er, Erb, (long long)S * 64);

  const float scaling = 0.125f;  // hd^-0.5 = 64^-0.5

  // QKV projection: bf16 out, q columns (col < E) scaled after bias
  gemm_bf16_wmma<<<dim3(3 * E / 128, NR / 128), 256, 0, stream>>>(
      X0b, Wqkv, ipb, nullptr, QKVb, E, 3 * E, scaling, E, 0);

  v_transpose<<<dim3(32 * 64 * S / 256), 256, 0, stream>>>(QKVb, Vtb);

  attn_rpr_flash<<<dim3(S / 16, 32), 32, 0, stream>>>(QKVb, Erb, Vtb, AOb);

  gemm_bf16_wmma<<<dim3(E / 128, NR / 128), 256, 0, stream>>>(
      AOb, Wout, opb, PROJ, nullptr, E, E, 1.f, 0, 0);

  add_layernorm<<<dim3(NR), 256, 0, stream>>>(src, PROJ, g1, be1, X1f, X1b, E);

  gemm_bf16_wmma<<<dim3(F / 128, NR / 128), 256, 0, stream>>>(
      X1b, W1b, b1, nullptr, Hb, E, F, 1.f, 0, 1);

  gemm_bf16_wmma<<<dim3(E / 128, NR / 128), 256, 0, stream>>>(
      Hb, W2b, b2, FFf, nullptr, F, E, 1.f, 0, 0);

  add_layernorm<<<dim3(NR), 256, 0, stream>>>(X1f, FFf, g2, be2, out, nullptr, E);
}
